// GNN_8332236554306
// MI455X (gfx1250) — matrix operand order repaired and checked by reading, not compiled, not run on hardware
//
#include <hip/hip_runtime.h>
#include <hip/hip_bf16.h>

// ---------------------------------------------------------------------------
// GNN: 2x GraphConv (sum aggr) + ReLU + global mean pool + linear head.
// N=100000, E=1600000, F=H=128, G=64.
// Scatter phase: f32 hardware atomics (L2-resident, dominates runtime).
// GEMM phase:   [agg|x](Nx256) @ Wcat(256x128) via v_wmma_f32_16x16x32_f16.
// ---------------------------------------------------------------------------

typedef __attribute__((ext_vector_type(16))) _Float16 v16h;
typedef __attribute__((ext_vector_type(8)))  float    v8f;

#define FEAT 128
#define KTOT 256            // concat K: [agg | x]
#define KSTEPS 8            // 256 / 32
#define CTILES 8            // 128 / 16 feature column tiles
#define LDS_PAD 8
#define LDS_ROW (KTOT + LDS_PAD)   // 264 halves -> 528B row stride (16B aligned)

// ---------------- zero fill (float4 stores) ----------------
__global__ void gnn_zero_f32(float* __restrict__ p, long long n4) {
  long long i = (long long)blockIdx.x * blockDim.x + threadIdx.x;
  if (i < n4) ((float4*)p)[i] = make_float4(0.f, 0.f, 0.f, 0.f);
}

// ---------------- pack [W_rel; W_root] (f32, KxN row-major) into f16
// WMMA B-fragment order: fragIdx = ((s*8 + j)*32 + lane)*16 + h
// lane: n = j*16 + (lane&15), khalf = lane>>4
// h (0..15): klocal = khalf*8 + (h&7) + (h>>3)*16 (ISA table order) ; k = s*32+klocal
__global__ void gnn_pack_w(const float* __restrict__ Wrel,
                           const float* __restrict__ Wroot,
                           _Float16* __restrict__ packed) {
  int t = blockIdx.x * blockDim.x + threadIdx.x;       // 0..32767
  if (t >= KSTEPS * CTILES * 32 * 16) return;
  int h = t & 15, l = (t >> 4) & 31, j = (t >> 9) & 7, s = t >> 12;
  int khalf = l >> 4, n16 = l & 15;
  int klocal = khalf * 8 + (h & 7) + (h >> 3) * 16;
  int k = s * 32 + klocal;
  int n = j * 16 + n16;
  float w = (k < FEAT) ? Wrel[k * FEAT + n] : Wroot[(k - FEAT) * FEAT + n];
  packed[t] = (_Float16)w;
}

// ---------------- per-graph node counts ----------------
__global__ void gnn_count(const long long* __restrict__ batch, float* __restrict__ cnt, int n) {
  int i = blockIdx.x * blockDim.x + threadIdx.x;
  if (i < n) unsafeAtomicAdd(&cnt[batch[i]], 1.0f);
}

// ---------------- edge scatter: agg[dst] += w * x[src] ----------------
// 32 lanes per edge, float4 per lane, 4 hw float atomics.
__global__ __launch_bounds__(256)
void gnn_scatter(const float* __restrict__ xin, const long long* __restrict__ ei,
                 const float* __restrict__ ew, float* __restrict__ agg, long long nE) {
  long long gid = (long long)blockIdx.x * blockDim.x + threadIdx.x;
  long long e = gid >> 5;
  int lane = gid & 31;
  if (e >= nE) return;
  long long src = ei[e];
  long long dst = ei[nE + e];
  float w = ew[e];
  float4 v = ((const float4*)(xin + src * FEAT))[lane];
  float* ad = agg + dst * FEAT + lane * 4;
  unsafeAtomicAdd(ad + 0, w * v.x);
  unsafeAtomicAdd(ad + 1, w * v.y);
  unsafeAtomicAdd(ad + 2, w * v.z);
  unsafeAtomicAdd(ad + 3, w * v.w);
}

// ---------------- WMMA GEMM: out = act( [A0|A1] @ packedW + bias ) ----------
// Block: 256 threads = 8 waves, 16 nodes. Wave j owns feature cols j*16..+15.
// A fragment per ISA 16-bit A 16x32 layout; B fragment pre-packed contiguous.
__global__ __launch_bounds__(256)
void gnn_gemm(const float* __restrict__ A0,          // K 0..127   (agg)
              const float* __restrict__ A1,          // K 128..255 (x or h)
              const _Float16* __restrict__ packedW,  // 256x128 f16, frag order
              const float* __restrict__ bias,        // [128]
              float* __restrict__ outH,              // [N,128] (if !doPool)
              const long long* __restrict__ batch,   // [N]     (if doPool)
              float* __restrict__ pooled,            // [G,128] (if doPool)
              int doRelu, int doPool, int nNodes) {
  __shared__ _Float16 lsA[16][LDS_ROW];
  const int nodeBase = blockIdx.x * 16;
  const int tid = threadIdx.x;

  // Stage 16x256 f32 panel -> f16 LDS (4 float4 per thread).
#pragma unroll
  for (int i = 0; i < 4; ++i) {
    int idx = tid + 256 * i;            // 0..1023 float4 slots
    int nd  = idx >> 6;                 // node within tile
    int f   = (idx & 63) * 4;           // concat-K offset
    int gn  = nodeBase + nd;
    float4 v = make_float4(0.f, 0.f, 0.f, 0.f);
    if (gn < nNodes) {
      const float* sp = (f < FEAT) ? (A0 + (long long)gn * FEAT + f)
                                   : (A1 + (long long)gn * FEAT + (f - FEAT));
      v = *(const float4*)sp;
    }
    lsA[nd][f + 0] = (_Float16)v.x;
    lsA[nd][f + 1] = (_Float16)v.y;
    lsA[nd][f + 2] = (_Float16)v.z;
    lsA[nd][f + 3] = (_Float16)v.w;
  }
  __syncthreads();

  const int wave  = tid >> 5;           // column tile j = 0..7
  const int lane  = tid & 31;
  const int m     = lane & 15;
  const int khalf = lane >> 4;

  v8f c = {};
#pragma unroll
  for (int s = 0; s < KSTEPS; ++s) {
    // A: halves 0..7 = K klocal khalf*8+0..7 ; halves 8..15 = +16 (ISA table)
    union { uint4 u[2]; v16h v; } au;
    au.u[0] = *(const uint4*)&lsA[m][s * 32 + khalf * 8];
    au.u[1] = *(const uint4*)&lsA[m][s * 32 + khalf * 8 + 16];
    // B: one contiguous 32B fragment per lane
    v16h b = *(const v16h*)(packedW + (((size_t)(s * CTILES + wave) * 32 + lane) * 16));
    c = __builtin_amdgcn_wmma_f32_16x16x32_f16(
        /*neg_a=*/false, au.v, /*neg_b=*/false, b,
        /*c_mod=*/(short)0, c, /*reuse_a=*/false, /*reuse_b=*/false);
  }

  // Epilogue. D layout: lane n = col, VGPR r -> row khalf*8 + r.
  const int nfeat = wave * 16 + (lane & 15);
  const float bval = bias[nfeat];
#pragma unroll
  for (int r = 0; r < 8; ++r) {
    int gn = nodeBase + khalf * 8 + r;
    if (gn >= nNodes) continue;
    float val = c[r] + bval;
    if (doRelu) val = fmaxf(val, 0.f);
    if (doPool) {
      long long g = batch[gn];
      unsafeAtomicAdd(&pooled[g * FEAT + nfeat], val);
    } else {
      outH[(long long)gn * FEAT + nfeat] = val;
    }
  }
}

// ---------------- head: out[g] = (pooled[g]/cnt[g]) . W_lin + b_lin --------
__global__ void gnn_head(const float* __restrict__ pooled, const float* __restrict__ cnt,
                         const float* __restrict__ Wlin, const float* __restrict__ blin,
                         float* __restrict__ out, int nG) {
  int g = blockIdx.x * blockDim.x + threadIdx.x;
  if (g >= nG) return;
  float inv = 1.0f / fmaxf(cnt[g], 1.0f);
  float acc = 0.f;
#pragma unroll 4
  for (int f = 0; f < FEAT; ++f) acc += pooled[g * FEAT + f] * Wlin[f];
  out[g] = acc * inv + blin[0];
}

// ---------------------------------------------------------------------------
extern "C" void kernel_launch(void* const* d_in, const int* in_sizes, int n_in,
                              void* d_out, int out_size, void* d_ws, size_t ws_size,
                              hipStream_t stream) {
  const float*     x      = (const float*)d_in[0];
  const long long* ei     = (const long long*)d_in[1];
  const float*     ew     = (const float*)d_in[2];
  const long long* batch  = (const long long*)d_in[3];
  const float*     Wrel1  = (const float*)d_in[4];
  const float*     brel1  = (const float*)d_in[5];
  const float*     Wroot1 = (const float*)d_in[6];
  const float*     Wrel2  = (const float*)d_in[7];
  const float*     brel2  = (const float*)d_in[8];
  const float*     Wroot2 = (const float*)d_in[9];
  const float*     Wlin   = (const float*)d_in[10];
  const float*     blin   = (const float*)d_in[11];
  float*           out    = (float*)d_out;

  const int       nN = in_sizes[0] / FEAT;     // 100000
  const long long nE = (long long)in_sizes[2]; // 1600000
  const int       nG = out_size;               // 64

  // workspace carve-up (d_ws assumed >= ~103 MB)
  char* ws = (char*)d_ws;
  _Float16* packedW1 = (_Float16*)ws;                       // 64 KB
  _Float16* packedW2 = (_Float16*)(ws + 65536);             // 64 KB
  float* agg    = (float*)(ws + 131072);                    // N*128 f32
  float* h      = agg + (size_t)nN * FEAT;                  // N*128 f32
  float* pooled = h + (size_t)nN * FEAT;                    // G*128 f32
  float* cnt    = pooled + (size_t)nG * FEAT;               // G f32

  const long long aggN4  = (long long)nN * FEAT / 4;
  const long long poolN4 = (long long)nG * FEAT / 4;
  dim3 blk(256);
  auto blocks4 = [](long long n4) { return dim3((unsigned)((n4 + 255) / 256)); };

  // init
  gnn_zero_f32<<<blocks4(aggN4), blk, 0, stream>>>(agg, aggN4);
  gnn_zero_f32<<<blocks4(poolN4), blk, 0, stream>>>(pooled, poolN4);
  gnn_zero_f32<<<blocks4((nG + 3) / 4), blk, 0, stream>>>(cnt, (nG + 3) / 4);
  gnn_pack_w<<<dim3(128), blk, 0, stream>>>(Wrel1, Wroot1, packedW1);
  gnn_pack_w<<<dim3(128), blk, 0, stream>>>(Wrel2, Wroot2, packedW2);
  gnn_count<<<dim3((nN + 255) / 256), blk, 0, stream>>>(batch, cnt, nN);

  const dim3 scatterGrid((unsigned)((nE * 32 + 255) / 256));
  const dim3 gemmGrid((unsigned)((nN + 15) / 16));

  // layer 1: agg = scatter(x); h = relu([agg|x] @ Wcat1 + b1)
  gnn_scatter<<<scatterGrid, blk, 0, stream>>>(x, ei, ew, agg, nE);
  gnn_gemm<<<gemmGrid, blk, 0, stream>>>(agg, x, packedW1, brel1,
                                         h, batch, pooled, /*relu=*/1, /*pool=*/0, nN);

  // layer 2: agg = scatter(h); pooled += [agg|h] @ Wcat2 + b2 (fused pool)
  gnn_zero_f32<<<blocks4(aggN4), blk, 0, stream>>>(agg, aggN4);
  gnn_scatter<<<scatterGrid, blk, 0, stream>>>(h, ei, ew, agg, nE);
  gnn_gemm<<<gemmGrid, blk, 0, stream>>>(agg, h, packedW2, brel2,
                                         (float*)nullptr, batch, pooled, /*relu=*/0, /*pool=*/1, nN);

  // head
  gnn_head<<<dim3(1), dim3(64), 0, stream>>>(pooled, cnt, Wlin, blin, out, nG);
}
